// LSSTransform_38388417692551
// MI455X (gfx1250) — matrix operand order repaired and checked by reading, not compiled, not run on hardware
//
#include <hip/hip_runtime.h>
#include <hip/hip_bf16.h>

typedef __attribute__((ext_vector_type(16))) _Float16 v16h;
typedef __attribute__((ext_vector_type(8)))  _Float16 v8h;
typedef __attribute__((ext_vector_type(8)))  float    v8f;

// Problem constants
#define NIMG   12      // B*N
#define CIN    512
#define MOUT   123     // DBINS + C_TR
#define MPAD   128
#define NPIX   704     // FH*FW = 16*44
#define FHh    16
#define FWw    44
#define DBINS  59
#define CTR    64
#define NXg    128
#define NYg    128
#define NZg    1
#define BATCH  2

// Workspace layout (bytes)
#define OFF_COMB  ((size_t)0)                                   // 12*9 f32
#define OFF_WA    ((size_t)512)                                 // 128*512 f16
#define OFF_WB    (OFF_WA + (size_t)MPAD*CIN*2)                 // 12*704*512 f16
#define OFF_FEAT  (OFF_WB + (size_t)NIMG*NPIX*CIN*2)           // 12*704*128 f32

// ---------------------------------------------------------------------------
// 0) zero the output accumulator
__global__ void zero_out_kernel(float* out, int n) {
    int i = blockIdx.x * blockDim.x + threadIdx.x;
    if (i < n) out[i] = 0.0f;
}

// ---------------------------------------------------------------------------
// 1) combine = rots @ inv(intrins), 12 small 3x3 problems
__global__ void setup_combine_kernel(const float* __restrict__ rots,
                                     const float* __restrict__ intrins,
                                     float* __restrict__ comb) {
    int t = threadIdx.x;
    if (t >= NIMG) return;
    const float* K = intrins + t * 9;
    const float* R = rots    + t * 9;
    float a=K[0],b=K[1],c=K[2],d=K[3],e=K[4],f=K[5],g=K[6],h=K[7],i=K[8];
    float A  = e*i - f*h;
    float Bc = -(d*i - f*g);
    float Cc = d*h - e*g;
    float invdet = 1.0f / (a*A + b*Bc + c*Cc);
    float inv[9];
    inv[0] = A*invdet;          inv[1] = (c*h - b*i)*invdet; inv[2] = (b*f - c*e)*invdet;
    inv[3] = Bc*invdet;         inv[4] = (a*i - c*g)*invdet; inv[5] = (c*d - a*f)*invdet;
    inv[6] = Cc*invdet;         inv[7] = (b*g - a*h)*invdet; inv[8] = (a*e - b*d)*invdet;
    #pragma unroll
    for (int ii = 0; ii < 3; ++ii)
        #pragma unroll
        for (int jj = 0; jj < 3; ++jj)
            comb[t*9 + ii*3 + jj] =
                R[ii*3+0]*inv[0*3+jj] + R[ii*3+1]*inv[1*3+jj] + R[ii*3+2]*inv[2*3+jj];
}

// ---------------------------------------------------------------------------
// 2) w_depth f32 (123x512) -> f16 (128x512, zero padded rows)
__global__ void convert_w_kernel(const float* __restrict__ w, _Float16* __restrict__ wa) {
    int idx = blockIdx.x * blockDim.x + threadIdx.x;
    if (idx >= MPAD * CIN) return;
    int row = idx >> 9;
    int col = idx & (CIN - 1);
    wa[idx] = (row < MOUT) ? (_Float16)w[row * CIN + col] : (_Float16)0.0f;
}

// ---------------------------------------------------------------------------
// 3) x f32 (img, c, p) -> f16 (img, p, c)   LDS-tiled transpose
__global__ void convert_x_kernel(const float* __restrict__ x, _Float16* __restrict__ bt) {
    __shared__ float tile[32][33];
    int img = blockIdx.z;
    int p0  = blockIdx.x * 32;
    int c0  = blockIdx.y * 32;
    #pragma unroll
    for (int j = 0; j < 4; ++j) {
        int c = c0 + threadIdx.y + j * 8;
        int p = p0 + threadIdx.x;
        tile[threadIdx.y + j * 8][threadIdx.x] = x[((size_t)img * CIN + c) * NPIX + p];
    }
    __syncthreads();
    #pragma unroll
    for (int j = 0; j < 4; ++j) {
        int p = p0 + threadIdx.y + j * 8;
        int c = c0 + threadIdx.x;
        bt[((size_t)img * NPIX + p) * CIN + c] =
            (_Float16)tile[threadIdx.x][threadIdx.y + j * 8];
    }
}

// ---------------------------------------------------------------------------
// 4) WMMA GEMM: feat[img][pix][ch] = W(128x512) x X(512x704) + bias
//    block = 256 threads = 8 waves; wave w owns M rows [16w,16w+16); tile N=64.
__global__ void gemm_wmma_kernel(const _Float16* __restrict__ wa,   // [128][512]
                                 const _Float16* __restrict__ bt,   // [img][704][512]
                                 const float*    __restrict__ bias, // [123]
                                 float* __restrict__ feat) {        // [img][704][128]
    const int img   = blockIdx.x / (NPIX / 64);
    const int ptile = blockIdx.x % (NPIX / 64);
    const int wave  = threadIdx.x >> 5;
    const int lane  = threadIdx.x & 31;
    const int lhi   = lane >> 4;       // 0: lanes 0-15, 1: lanes 16-31
    const int l16   = lane & 15;

    // A fragment addressing (16-bit A 16x32, this code's contiguous K order: 16 K per lane)
    const int arow  = wave * 16 + l16;
    const int aoff  = lhi * 16;         // lane>=16 starts at K+16
    const _Float16* abase = wa + (size_t)arow * CIN;

    // B fragment addressing (16-bit B 32x16, this code's order: contiguous 16 K per lane, same as A)
    const int boff  = lhi * 16;         // lanes>=16 hold K+16..K+31
    const _Float16* bimg = bt + (size_t)img * NPIX * CIN;

    // bias vector for this lane's output channels
    const int chb = (wave << 4) + (lhi << 3);
    float bias8[8];
    #pragma unroll
    for (int r = 0; r < 8; ++r) {
        int ch = chb + r;
        bias8[r] = (ch < MOUT) ? bias[ch] : 0.0f;
    }

    v8f acc[4] = {v8f{}, v8f{}, v8f{}, v8f{}};

    for (int k0 = 0; k0 < CIN; k0 += 32) {
        // A fragment: two contiguous 16B chunks per lane
        v8h alo = *(const v8h*)(abase + k0 + aoff);
        v8h ahi = *(const v8h*)(abase + k0 + aoff + 8);
        v16h afrag;
        #pragma unroll
        for (int i = 0; i < 8; ++i) { afrag[i] = alo[i]; afrag[i + 8] = ahi[i]; }
        __builtin_prefetch(abase + k0 + 32, 0, 1);

        #pragma unroll
        for (int nt = 0; nt < 4; ++nt) {
            int pix = ptile * 64 + nt * 16 + l16;
            v16h bfrag = *(const v16h*)(bimg + (size_t)pix * CIN + k0 + boff);
            acc[nt] = __builtin_amdgcn_wmma_f32_16x16x32_f16(
                false, afrag, false, bfrag, (short)0, acc[nt], false, false);
        }
    }

    // D layout: VGPR r, lanes 0-15 -> M=r, lanes 16-31 -> M=r+8; N = lane%16
    #pragma unroll
    for (int nt = 0; nt < 4; ++nt) {
        int pix = ptile * 64 + nt * 16 + l16;
        v8f v = acc[nt];
        #pragma unroll
        for (int r = 0; r < 8; ++r) v[r] += bias8[r];
        *(v8f*)(feat + ((size_t)img * NPIX + pix) * MPAD + chb) = v;
    }
}

// ---------------------------------------------------------------------------
// 5) per-pixel softmax over 59 depth logits + voxel scatter of depth*ctx
//    one wave (32 lanes) per pixel; 8 pixels per 256-thread block
__global__ void softmax_scatter_kernel(const float* __restrict__ feat,   // [img][704][128]
                                       const float* __restrict__ comb,   // [12][9]
                                       const float* __restrict__ trans,  // [12][3]
                                       float* __restrict__ out) {        // [B][64][128][128]
    __shared__ float sd[8][64];   // depth probs per wave
    __shared__ float sc[8][64];   // ctx features per wave
    const int wave = threadIdx.x >> 5;
    const int lane = threadIdx.x & 31;
    const int pixglob = blockIdx.x * 8 + wave;          // 1056*8 == 12*704 exactly
    const int img = pixglob / NPIX;
    const int p   = pixglob % NPIX;
    const int h   = p / FWw;
    const int w   = p % FWw;
    const int b   = img / 6;

    const float* fbase = feat + (size_t)pixglob * MPAD;
    float f0 = fbase[lane];
    float f1 = fbase[lane + 32];
    float f2 = fbase[lane + 64];
    float f3 = fbase[lane + 96];

    // softmax over channels [0,59)
    float m = f0;                                   // ch = lane (0..31)
    if (lane < 27) m = fmaxf(m, f1);                // ch = lane+32 (32..58)
    #pragma unroll
    for (int off = 16; off > 0; off >>= 1) m = fmaxf(m, __shfl_xor(m, off, 32));
    float e0 = __expf(f0 - m);
    float e1 = (lane < 27) ? __expf(f1 - m) : 0.0f;
    float s = e0 + e1;
    #pragma unroll
    for (int off = 16; off > 0; off >>= 1) s += __shfl_xor(s, off, 32);
    float rs = 1.0f / s;

    sd[wave][lane] = e0 * rs;
    if (lane < 27) sd[wave][lane + 32] = e1 * rs;
    // ctx channels 59..122 -> c = ch-59
    if (lane >= 27) sc[wave][lane - 27] = f1;       // c 0..4
    sc[wave][lane + 5] = f2;                        // c 5..36
    if (lane < 27) sc[wave][lane + 37] = f3;        // c 37..63
    __syncthreads();

    // wave-uniform geometry setup
    const float* cm = comb + img * 9;
    float c00 = cm[0], c01 = cm[1], c02 = cm[2];
    float c10 = cm[3], c11 = cm[4], c12 = cm[5];
    float c20 = cm[6], c21 = cm[7], c22 = cm[8];
    float t0 = trans[img*3 + 0], t1 = trans[img*3 + 1], t2 = trans[img*3 + 2];
    float xw = (float)w * (703.0f / 43.0f);
    float yh = (float)h * 17.0f;

    for (int d = 0; d < DBINS; ++d) {
        float dv = 1.0f + (float)d;
        float px = xw * dv, py = yh * dv, pz = dv;
        float gx = c00*px + c01*py + c02*pz + t0;
        float gy = c10*px + c11*py + c12*pz + t1;
        float gz = c20*px + c21*py + c22*pz + t2;
        int ix = (int)((gx + 51.2f) * 1.25f);     // trunc toward zero == jnp.trunc
        int iy = (int)((gy + 51.2f) * 1.25f);
        int iz = (int)((gz + 10.0f) * 0.05f);
        // match reference kept-check on trunc'd indices
        bool kept = (gx + 51.2f) > -0.8f && ix >= 0 && ix < NXg &&
                    (gy + 51.2f) > -0.8f && iy >= 0 && iy < NYg &&
                    (gz + 10.0f) > -20.f && iz >= 0 && iz < NZg;
        // note: negative fractions trunc to 0 and ARE kept in the reference;
        // the (int) cast reproduces that, extra guards only reject <= -1 voxels
        if (kept) {
            float dw = sd[wave][d];
            size_t base = ((size_t)b * CTR * NXg * NYg) + (size_t)ix * NYg + iy;
            int c = lane;
            atomicAdd(&out[base + (size_t)c * (NXg * NYg)], dw * sc[wave][c]);
            c = lane + 32;
            atomicAdd(&out[base + (size_t)c * (NXg * NYg)], dw * sc[wave][c]);
        }
    }
}

// ---------------------------------------------------------------------------
extern "C" void kernel_launch(void* const* d_in, const int* in_sizes, int n_in,
                              void* d_out, int out_size, void* d_ws, size_t ws_size,
                              hipStream_t stream) {
    const float* x       = (const float*)d_in[0];   // [2,6,512,16,44]
    const float* rots    = (const float*)d_in[1];   // [2,6,3,3]
    const float* trans   = (const float*)d_in[2];   // [2,6,3]
    const float* intrins = (const float*)d_in[3];   // [2,6,3,3]
    const float* w_depth = (const float*)d_in[4];   // [123,512]
    const float* b_depth = (const float*)d_in[5];   // [123]
    float* out = (float*)d_out;

    char* ws = (char*)d_ws;
    float*    combWS = (float*)   (ws + OFF_COMB);
    _Float16* waWS   = (_Float16*)(ws + OFF_WA);
    _Float16* btWS   = (_Float16*)(ws + OFF_WB);
    float*    featWS = (float*)   (ws + OFF_FEAT);

    // 0) zero output accumulator
    zero_out_kernel<<<(out_size + 255) / 256, 256, 0, stream>>>(out, out_size);

    // 1) combine matrices
    setup_combine_kernel<<<1, 32, 0, stream>>>(rots, intrins, combWS);

    // 2) weights -> f16 padded
    convert_w_kernel<<<(MPAD * CIN + 255) / 256, 256, 0, stream>>>(w_depth, waWS);

    // 3) x -> f16, pixel-major (transpose)
    convert_x_kernel<<<dim3(NPIX / 32, CIN / 32, NIMG), dim3(32, 8), 0, stream>>>(x, btWS);

    // 4) WMMA GEMM + bias
    gemm_wmma_kernel<<<NIMG * (NPIX / 64), 256, 0, stream>>>(waWS, btWS, b_depth, featWS);

    // 5) softmax + BEV scatter
    softmax_scatter_kernel<<<(NIMG * NPIX) / 8, 256, 0, stream>>>(featWS, combWS, trans, out);
}
